// CausalSelfAttention_3547642987063
// MI455X (gfx1250) — matrix operand order repaired and checked by reading, not compiled, not run on hardware
//
#include <hip/hip_runtime.h>
#include <hip/hip_bf16.h>
#include <math.h>

typedef __attribute__((ext_vector_type(16))) __bf16 v16bf;
typedef __attribute__((ext_vector_type(8)))  __bf16 v8bf;
typedef __attribute__((ext_vector_type(8)))  float  v8f;

#define LN10000 9.210340371976184f
#define T_SEQ 2048
#define D_MODEL 2048
#define HD 128
#define NH 16
#define BM (2 * T_SEQ)          // 4096 rows (B*T)

__device__ __forceinline__ v8f wmma_bf16(v16bf a, v16bf b, v8f c) {
  return __builtin_amdgcn_wmma_f32_16x16x32_bf16(false, a, false, b, (short)0, c,
                                                 false, false);
}

// A-fragment (16x32 bf16): lane = row (lane&15); elems 0..7 -> K = k0 + (lane/16)*8 + j,
// elems 8..15 -> K = k0 + 16 + (lane/16)*8 + (j-8).  Two contiguous 16B loads.
__device__ __forceinline__ v16bf load_a_frag(const __bf16* A, int lda, int row0,
                                             int k0, int lane) {
  const __bf16* p = A + (size_t)(row0 + (lane & 15)) * lda + k0 + ((lane >> 4) << 3);
  v8bf lo = *(const v8bf*)p;
  v8bf hi = *(const v8bf*)(p + 16);
  v16bf a;
#pragma unroll
  for (int i = 0; i < 8; ++i) { a[i] = lo[i]; a[i + 8] = hi[i]; }
  return a;
}

// B-fragment (32x16 bf16): lane = col (lane&15); elems 0..7 -> K = k0 + (lane/16)*8 + j,
// elems 8..15 -> K = k0 + 16 + (lane/16)*8 + (j-8).  Bm is stored [col][K] -> two 16B loads.
__device__ __forceinline__ v16bf load_b_frag(const __bf16* Bm, int ldb, int col0,
                                             int k0, int lane) {
  const __bf16* p = Bm + (size_t)(col0 + (lane & 15)) * ldb + k0 + ((lane >> 4) << 3);
  v8bf lo = *(const v8bf*)p;
  v8bf hi = *(const v8bf*)(p + 16);
  v16bf b;
#pragma unroll
  for (int i = 0; i < 8; ++i) { b[i] = lo[i]; b[i + 8] = hi[i]; }
  return b;
}

// Rotate one RoPE pair (values in/out through locals).
__device__ __forceinline__ void rope_rot(float& x1, float& x2, int dim, float trow) {
  float inv = __expf(-LN10000 * ((float)dim * (1.0f / (float)HD)));
  float s, c;
  __sincosf(trow * inv, &s, &c);
  float r1 = x1 * c - x2 * s;
  float r2 = x1 * s + x2 * c;
  x1 = r1;
  x2 = r2;
}

__global__ void cvt_bf16_kernel(const float* __restrict__ in, __bf16* __restrict__ out,
                                int n) {
  int i = blockIdx.x * blockDim.x + threadIdx.x;
  if (i < n) out[i] = (__bf16)in[i];
}

// In-place RoPE over a [b,h,t,d] bf16 tensor; one thread = one (row, pair).
__global__ void rope_kernel(__bf16* __restrict__ qk, int totalPairs) {
  int i = blockIdx.x * blockDim.x + threadIdx.x;
  if (i >= totalPairs) return;
  int pair = i & (HD / 2 - 1);     // 0..63
  int row = i >> 6;                // bh*T + t
  int t = row & (T_SEQ - 1);
  size_t off = (size_t)row * HD + 2 * pair;
  float x1 = (float)qk[off], x2 = (float)qk[off + 1];
  rope_rot(x1, x2, 2 * pair, (float)t);
  qk[off] = (__bf16)x1;
  qk[off + 1] = (__bf16)x2;
}

// qkv = x @ w_attn.T ; scatter into qh/kh [b,h,t,d] (bf16) and vT [b,h,d,t] (bf16).
__global__ void gemm_qkv_kernel(const __bf16* __restrict__ xb,
                                const __bf16* __restrict__ wab,
                                __bf16* __restrict__ qh, __bf16* __restrict__ kh,
                                __bf16* __restrict__ vT) {
  const int lane = threadIdx.x & 31;
  const int wave = (blockIdx.x * blockDim.x + threadIdx.x) >> 5;
  const int nTiles = (3 * D_MODEL) / 64;   // 96
  const int mBase = (wave / nTiles) * 16;
  const int nBase = (wave % nTiles) * 64;

  v8f zero = {};
  v8f acc[4] = {zero, zero, zero, zero};
  for (int k = 0; k < D_MODEL; k += 32) {
    v16bf a = load_a_frag(xb, D_MODEL, mBase, k, lane);
    __builtin_prefetch(xb + (size_t)(mBase + (lane & 15)) * D_MODEL + k + 64, 0, 1);
#pragma unroll
    for (int j = 0; j < 4; ++j)
      acc[j] = wmma_bf16(a, load_b_frag(wab, D_MODEL, nBase + 16 * j, k, lane), acc[j]);
  }

  const int hf = lane >> 4, colL = lane & 15;
#pragma unroll
  for (int j = 0; j < 4; ++j) {
    int n = nBase + 16 * j + colL;
    int region = n >> 11;    // 0=q 1=k 2=v (uniform per wave/j)
    int nn = n & (D_MODEL - 1);
    int h = nn >> 7;
    int d = nn & (HD - 1);
#pragma unroll
    for (int r = 0; r < 8; ++r) {
      int m = mBase + hf * 8 + r;
      int b_ = m >> 11;
      int t = m & (T_SEQ - 1);
      __bf16 v = (__bf16)acc[j][r];
      size_t bh = (size_t)b_ * NH + h;
      if (region == 0)      qh[(bh * T_SEQ + t) * HD + d] = v;
      else if (region == 1) kh[(bh * T_SEQ + t) * HD + d] = v;
      else                  vT[(bh * HD + d) * T_SEQ + t] = v;
    }
  }
}

// Flash-style causal attention: one wave = 16 query rows of one (b,h).
// RoPE already applied to qh/kh, so the hot loop is loads + WMMA + softmax only.
__global__ void attn_kernel(const __bf16* __restrict__ qh, const __bf16* __restrict__ kh,
                            const __bf16* __restrict__ vT, __bf16* __restrict__ yb) {
  __shared__ __align__(32) __bf16 plds[4][16 * 32];
  const int lane = threadIdx.x & 31;
  const int wIb = threadIdx.x >> 5;
  const int wave = blockIdx.x * 4 + wIb;       // 0..4095
  const int qTile = wave & 127;
  const int bh = wave >> 7;                    // b*NH + h
  const int qBase = qTile * 16;
  const __bf16* Q = qh + (size_t)bh * T_SEQ * HD;
  const __bf16* K = kh + (size_t)bh * T_SEQ * HD;
  const __bf16* V = vT + (size_t)bh * HD * T_SEQ;
  __bf16* pbuf = &plds[wIb][0];

  const int hf = lane >> 4, colL = lane & 15;

  v16bf qf[4];
#pragma unroll
  for (int kk = 0; kk < 4; ++kk)
    qf[kk] = load_a_frag(Q, HD, qBase, kk * 32, lane);

  v8f zero = {};
  v8f o[8];
  float mrun[8], lrun[8];
#pragma unroll
  for (int i = 0; i < 8; ++i) { o[i] = zero; mrun[i] = -1e30f; lrun[i] = 0.0f; }

  const float scale = 0.08838834764831845f;    // 1/sqrt(128)

  for (int jBase = 0; jBase <= qBase + 15; jBase += 32) {
    v8f s0 = zero, s1 = zero;
#pragma unroll
    for (int kk = 0; kk < 4; ++kk) {
      s0 = wmma_bf16(qf[kk], load_b_frag(K, HD, jBase, kk * 32, lane), s0);
      s1 = wmma_bf16(qf[kk], load_b_frag(K, HD, jBase + 16, kk * 32, lane), s1);
    }
    // online softmax per row (row is spread over a 16-lane half at element r)
#pragma unroll
    for (int r = 0; r < 8; ++r) {
      int t = qBase + hf * 8 + r;
      float v0 = s0[r] * scale, v1 = s1[r] * scale;
      if (jBase + colL > t)      v0 = -1e30f;
      if (jBase + 16 + colL > t) v1 = -1e30f;
      float mx = fmaxf(v0, v1);
#pragma unroll
      for (int off = 1; off < 16; off <<= 1)
        mx = fmaxf(mx, __shfl_xor(mx, off, 32));
      float mnew = fmaxf(mrun[r], mx);
      float alpha = __expf(mrun[r] - mnew);
      float p0 = __expf(v0 - mnew);
      float p1 = __expf(v1 - mnew);
      float ps = p0 + p1;
#pragma unroll
      for (int off = 1; off < 16; off <<= 1)
        ps += __shfl_xor(ps, off, 32);
      lrun[r] = lrun[r] * alpha + ps;
      mrun[r] = mnew;
#pragma unroll
      for (int dt = 0; dt < 8; ++dt) o[dt][r] *= alpha;
      pbuf[(hf * 8 + r) * 32 + colL] = (__bf16)p0;
      pbuf[(hf * 8 + r) * 32 + 16 + colL] = (__bf16)p1;
    }
    asm volatile("s_wait_dscnt 0" ::: "memory");  // P tile visible before re-read
    v16bf pa = load_a_frag(pbuf, 32, 0, 0, lane);
#pragma unroll
    for (int dt = 0; dt < 8; ++dt)
      o[dt] = wmma_bf16(pa, load_b_frag(V, T_SEQ, dt * 16, jBase, lane), o[dt]);
  }

  // normalize & store yb[b*T + t][h*128 + d]
  const int b_ = bh >> 4, h = bh & 15;
#pragma unroll
  for (int r = 0; r < 8; ++r) {
    float inv = 1.0f / lrun[r];
    int t = qBase + hf * 8 + r;
    size_t row = ((size_t)b_ * T_SEQ + t) * D_MODEL + h * HD;
#pragma unroll
    for (int dt = 0; dt < 8; ++dt)
      yb[row + dt * 16 + colL] = (__bf16)(o[dt][r] * inv);
  }
}

// out = y @ w_proj.T  (f32 output)
__global__ void gemm_proj_kernel(const __bf16* __restrict__ yb,
                                 const __bf16* __restrict__ wpb,
                                 float* __restrict__ out) {
  const int lane = threadIdx.x & 31;
  const int wave = (blockIdx.x * blockDim.x + threadIdx.x) >> 5;
  const int nTiles = D_MODEL / 64;             // 32
  const int mBase = (wave / nTiles) * 16;
  const int nBase = (wave % nTiles) * 64;

  v8f zero = {};
  v8f acc[4] = {zero, zero, zero, zero};
  for (int k = 0; k < D_MODEL; k += 32) {
    v16bf a = load_a_frag(yb, D_MODEL, mBase, k, lane);
#pragma unroll
    for (int j = 0; j < 4; ++j)
      acc[j] = wmma_bf16(a, load_b_frag(wpb, D_MODEL, nBase + 16 * j, k, lane), acc[j]);
  }
  const int hf = lane >> 4, colL = lane & 15;
#pragma unroll
  for (int j = 0; j < 4; ++j) {
    int n = nBase + 16 * j + colL;
#pragma unroll
    for (int r = 0; r < 8; ++r) {
      int m = mBase + hf * 8 + r;
      out[(size_t)m * D_MODEL + n] = acc[j][r];
    }
  }
}

extern "C" void kernel_launch(void* const* d_in, const int* in_sizes, int n_in,
                              void* d_out, int out_size, void* d_ws, size_t ws_size,
                              hipStream_t stream) {
  const float* x      = (const float*)d_in[0];
  const float* w_attn = (const float*)d_in[1];
  const float* w_proj = (const float*)d_in[2];
  float* out = (float*)d_out;

  __bf16* xb  = (__bf16*)d_ws;                       // 4096*2048
  __bf16* wab = xb  + (size_t)BM * D_MODEL;          // 6144*2048
  __bf16* wpb = wab + (size_t)3 * D_MODEL * D_MODEL; // 2048*2048
  __bf16* qhb = wpb + (size_t)D_MODEL * D_MODEL;     // 2*16*2048*128
  __bf16* khb = qhb + (size_t)BM * D_MODEL;
  __bf16* vtb = khb + (size_t)BM * D_MODEL;
  __bf16* ybb = vtb + (size_t)BM * D_MODEL;          // 4096*2048

  const int nX = BM * D_MODEL;            // 8388608
  const int nWA = 3 * D_MODEL * D_MODEL;  // 12582912
  const int nWP = D_MODEL * D_MODEL;      // 4194304
  cvt_bf16_kernel<<<(nX + 255) / 256, 256, 0, stream>>>(x, xb, nX);
  cvt_bf16_kernel<<<(nWA + 255) / 256, 256, 0, stream>>>(w_attn, wab, nWA);
  cvt_bf16_kernel<<<(nWP + 255) / 256, 256, 0, stream>>>(w_proj, wpb, nWP);

  // QKV GEMM: 256 m-tiles * 96 n-tiles = 24576 waves, 8 waves/block
  gemm_qkv_kernel<<<3072, 256, 0, stream>>>(xb, wab, qhb, khb, vtb);

  // One-shot RoPE on Q and K (in place), pairs adjacent in memory
  const int nPairs = BM * D_MODEL / 2;    // 4194304 per tensor
  rope_kernel<<<(nPairs + 255) / 256, 256, 0, stream>>>(qhb, nPairs);
  rope_kernel<<<(nPairs + 255) / 256, 256, 0, stream>>>(khb, nPairs);

  // Attention: 32 (b,h) * 128 q-tiles = 4096 waves, 4 waves/block
  attn_kernel<<<1024, 128, 0, stream>>>(qhb, khb, vtb, ybb);

  // Projection: 256 m-tiles * 32 n-tiles = 8192 waves, 8 waves/block
  gemm_proj_kernel<<<1024, 256, 0, stream>>>(ybb, wpb, out);
}
